// Attention_66108136620773
// MI455X (gfx1250) — compile-verified, matrix operand order checked by reading, not run on hardware
//
#include <hip/hip_runtime.h>
#include <hip/hip_bf16.h>

// ---------------- problem constants ----------------
#define B_   4
#define N_   2048
#define D_   1024
#define H_   16
#define DH_  64
#define M_   512
#define J_   2560            // N_ + M_
#define NR_  (B_*N_)         // 8192 total query rows

typedef __attribute__((ext_vector_type(16))) _Float16 v16h;
typedef __attribute__((ext_vector_type(8)))  float    v8f;
typedef __attribute__((ext_vector_type(4)))  unsigned int u32x4;
typedef __attribute__((ext_vector_type(8)))  unsigned int u32x8;

union HV16 { v16h v; uint4 q[2]; _Float16 h[16]; };

// Load a 16-half WMMA operand fragment: halves [0..7] from p[off..off+7],
// halves [8..15] from p[off+16..off+23]  (per CDNA5 16-bit A/B VGPR layout).
__device__ __forceinline__ v16h ld_frag(const _Float16* p, int off) {
    HV16 u;
    u.q[0] = *(const uint4*)(p + off);
    u.q[1] = *(const uint4*)(p + off + 16);
    return u.v;
}

__device__ __forceinline__ v8f wmma16(v16h a, v16h b, v8f c) {
    // D = A(16x32 f16) * B(32x16 f16) + C(16x16 f32)
    return __builtin_amdgcn_wmma_f32_16x16x32_f16(
        false, a, false, b, (short)0, c, false, false);
}

// Per-lane 16B async DMA: global -> LDS, tracked by ASYNCcnt.
__device__ __forceinline__ void async_copy16(unsigned int lds_off,
                                             const _Float16* gptr) {
    asm volatile("global_load_async_to_lds_b128 %0, %1, off"
                 :: "v"(lds_off), "v"((unsigned long long)(uintptr_t)gptr)
                 : "memory");
}

// ---------------- fp32 -> fp16 convert ----------------
__global__ void cvt_f16_kernel(const float* __restrict__ src,
                               _Float16* __restrict__ dst, int n) {
    int i = blockIdx.x * 256 + threadIdx.x;
    if (i < n) dst[i] = (_Float16)src[i];
}

// xl_memory (b,512,2,64) f32  ->  kh/vh rows [0,512) f16
__global__ void xl_splice_kernel(const float* __restrict__ xl,
                                 _Float16* __restrict__ kh,
                                 _Float16* __restrict__ vh) {
    int i = blockIdx.x * 256 + threadIdx.x;      // over 4*512*64
    if (i >= B_ * M_ * DH_) return;
    int d = i & 63, j = (i >> 6) & (M_ - 1), b = i >> 15;
    size_t src = (((size_t)b * M_ + j) * 2) * DH_ + d;
    size_t dst = ((size_t)b * J_ + j) * DH_ + d;
    kh[dst] = (_Float16)xl[src];
    vh[dst] = (_Float16)xl[src + DH_];
}

// ---------------- Q projection: qh(b,h,n,64) = x @ Wq * scale ----------------
__global__ __launch_bounds__(128)
void gemm_q_kernel(const _Float16* __restrict__ xh,
                   const _Float16* __restrict__ wqh,
                   _Float16* __restrict__ qh) {
    __shared__ _Float16 wt[64 * 32];             // (col, k) transposed chunk
    const int tid = threadIdx.x, lane = tid & 31, wv = tid >> 5;
    const int l16 = lane & 15, off = (lane < 16) ? 0 : 8;
    const int row0 = blockIdx.x * 64, n0 = blockIdx.y * 64;
    const _Float16* abase = xh + (size_t)(row0 + wv * 16 + l16) * D_;

    v8f c[4]; for (int t = 0; t < 4; ++t) c[t] = {};
    for (int k0 = 0; k0 < D_; k0 += 32) {
        __syncthreads();
        for (int it = tid; it < 256; it += 128) {        // 32x64 halves, 8 at a time
            int kk = it >> 3, cc = (it & 7) * 8;
            const _Float16* wp = &wqh[(size_t)(k0 + kk) * D_ + n0 + cc];
            for (int q = 0; q < 8; ++q) wt[(cc + q) * 32 + kk] = wp[q];
        }
        __syncthreads();
        v16h a = ld_frag(abase + k0, off);
        for (int t = 0; t < 4; ++t)
            c[t] = wmma16(a, ld_frag(&wt[(t * 16 + l16) * 32], off), c[t]);
    }
    const float scale = 0.125f;                  // 64^-0.5
    for (int t = 0; t < 4; ++t) {
        int col = n0 + t * 16 + l16, h = col >> 6, d = col & 63;
        for (int g = 0; g < 8; ++g) {
            int row = row0 + wv * 16 + off + g;
            int b = row >> 11, i = row & (N_ - 1);
            qh[(((size_t)b * H_ + h) * N_ + i) * DH_ + d] = (_Float16)(c[t][g] * scale);
        }
    }
}

// ---------------- KV projection + new_xl output ----------------
__global__ __launch_bounds__(128)
void gemm_kv_kernel(const _Float16* __restrict__ xh,
                    const _Float16* __restrict__ wkvh,
                    _Float16* __restrict__ kh, _Float16* __restrict__ vh,
                    float* __restrict__ out_xl) {
    __shared__ _Float16 wt[128 * 32];
    const int tid = threadIdx.x, lane = tid & 31, wv = tid >> 5;
    const int l16 = lane & 15, off = (lane < 16) ? 0 : 8;
    const int row0 = blockIdx.x * 64;
    const _Float16* abase = xh + (size_t)(row0 + wv * 16 + l16) * D_;

    v8f c[8]; for (int t = 0; t < 8; ++t) c[t] = {};
    for (int k0 = 0; k0 < D_; k0 += 32) {
        __syncthreads();
        for (int it = tid; it < 512; it += 128) {        // 32x128 halves
            int kk = it >> 4, cc = (it & 15) * 8;
            const _Float16* wp = &wkvh[(size_t)(k0 + kk) * 128 + cc];
            for (int q = 0; q < 8; ++q) wt[(cc + q) * 32 + kk] = wp[q];
        }
        __syncthreads();
        v16h a = ld_frag(abase + k0, off);
        for (int t = 0; t < 8; ++t)
            c[t] = wmma16(a, ld_frag(&wt[(t * 16 + l16) * 32], off), c[t]);
    }
    for (int t = 0; t < 8; ++t) {
        int col = t * 16 + l16, isv = col >> 6, d = col & 63;
        for (int g = 0; g < 8; ++g) {
            int row = row0 + wv * 16 + off + g;
            int b = row >> 11, i = row & (N_ - 1);
            float val = c[t][g];
            _Float16* dst = isv ? vh : kh;
            dst[((size_t)b * J_ + M_ + i) * DH_ + d] = (_Float16)val;
            if (i >= N_ - M_)                            // last 512 rows -> new_xl
                out_xl[(((size_t)b * M_ + (i - (N_ - M_))) * 2 + isv) * DH_ + d] = val;
        }
    }
}

// ---------------- flash attention ----------------
__global__ __launch_bounds__(128)
void attn_kernel(const _Float16* __restrict__ qh, const _Float16* __restrict__ kh,
                 const _Float16* __restrict__ vh, const float* __restrict__ bias,
                 _Float16* __restrict__ oh) {
    __shared__ _Float16 ks[32 * 64];             // K tile (key, dim) via async DMA
    __shared__ _Float16 vs[32 * 64];             // V tile (key, dim) via TDM
    __shared__ _Float16 pb[4 * 16 * 32];         // per-wave P scratch

    const int tid = threadIdx.x, lane = tid & 31, wv = tid >> 5;
    const int l16 = lane & 15, off = (lane < 16) ? 0 : 8;
    const int bx = blockIdx.x, h = blockIdx.y, b = blockIdx.z;
    const int r0 = bx * 64;                      // block query rows
    const int qr = r0 + wv * 16;                 // wave query rows

    const _Float16* qbase = qh + (((size_t)b * H_ + h) * N_ + qr + l16) * DH_;
    v16h aq0 = ld_frag(qbase + 0, off);          // head-dim K chunk 0..31
    v16h aq1 = ld_frag(qbase + 32, off);         // head-dim K chunk 32..63

    v8f o[4]; for (int t = 0; t < 4; ++t) o[t] = {};
    float mrow[8], lrow[8];
    for (int g = 0; g < 8; ++g) { mrow[g] = -3.0e38f; lrow[g] = 0.f; }

    int jend = r0 + 64 + M_; if (jend > J_) jend = J_;
    const _Float16* kg = kh + (size_t)b * J_ * DH_;
    const _Float16* vg = vh + (size_t)b * J_ * DH_;
    const unsigned int ks_base = (unsigned int)(uintptr_t)&ks[0];

    for (int j0 = 0; j0 < jend; j0 += 32) {
        __syncthreads();
        // ---- K tile: per-lane async DMA global->LDS (ASYNCcnt) ----
        for (int it = tid; it < 256; it += 128) {
            int key = it >> 3, doff = (it & 7) * 8;
            async_copy16(ks_base + (unsigned int)(key * 64 + doff) * 2u,
                         &kg[(size_t)(j0 + key) * DH_ + doff]);
        }
        // ---- V tile: Tensor Data Mover, one descriptor from wave 0 ----
        if (wv == 0) {
            unsigned long long ga =
                (unsigned long long)(uintptr_t)(vg + (size_t)j0 * DH_);
            unsigned int la = (unsigned int)(uintptr_t)&vs[0];
            u32x4 g0 = { 1u,                               // count=1, user mode
                         la,                               // lds_addr [63:32]
                         (unsigned int)ga,                 // global_addr lo
                         ((unsigned int)(ga >> 32) & 0x1FFFFFFu) | 0x80000000u };
            u32x8 g1 = { 0x10000u,     // data_size=2B (bits17:16=1), no mask
                         64u << 16,    // tensor_dim0[15:0]=64     (bits 79:48)
                         32u << 16,    // tensor_dim1[15:0]=32     (bits 111:80)
                         64u << 16,    // tile_dim0=64             (bits 127:112)
                         32u,          // tile_dim1=32 keys        (bits 143:128)
                         64u,          // tensor_dim0_stride=64    (bits 207:160)
                         0u, 0u };
            u32x4 gz = { 0u, 0u, 0u, 0u };
            asm volatile("tensor_load_to_lds %0, %1, %2, %3"
                         :: "s"(g0), "s"(g1), "s"(gz), "s"(gz)
                         : "memory");
            __builtin_amdgcn_s_wait_tensorcnt(0);
        }
        asm volatile("s_wait_asynccnt 0x0" ::: "memory");
        __syncthreads();

        // ---- S = Q * K^T : two 16x16 tiles ----
        v8f s[2];
        for (int u = 0; u < 2; ++u) {
            const _Float16* kr = &ks[(u * 16 + l16) * 64];
            v8f acc = {};
            acc = wmma16(aq0, ld_frag(kr + 0, off), acc);
            acc = wmma16(aq1, ld_frag(kr + 32, off), acc);
            s[u] = acc;
        }

        // ---- bias + causal mask ----
        const float* bp = bias + ((size_t)h * N_ + qr + off) * J_ + j0 + l16;
        __builtin_prefetch(bp + 32, 0, 0);               // global_prefetch_b8
        for (int g = 0; g < 8; ++g) {
            int row = qr + off + g, col = j0 + l16;
            float v0 = s[0][g] + bp[(size_t)g * J_];
            float v1 = s[1][g] + bp[(size_t)g * J_ + 16];
            if (col > row + M_)       v0 = -3.0e38f;
            if (col + 16 > row + M_)  v1 = -3.0e38f;
            s[0][g] = v0; s[1][g] = v1;
        }

        // ---- online softmax (rows live across 16 lanes) ----
        for (int g = 0; g < 8; ++g) {
            float t = fmaxf(s[0][g], s[1][g]);
            for (int msk = 8; msk >= 1; msk >>= 1)
                t = fmaxf(t, __shfl_xor(t, msk, 16));
            float mnew = fmaxf(mrow[g], t);
            float a = __expf(mrow[g] - mnew);
            mrow[g] = mnew;
            float p0 = __expf(s[0][g] - mnew);
            float p1 = __expf(s[1][g] - mnew);
            s[0][g] = p0; s[1][g] = p1;
            float rs = p0 + p1;
            for (int msk = 8; msk >= 1; msk >>= 1)
                rs += __shfl_xor(rs, msk, 16);
            lrow[g] = lrow[g] * a + rs;
            for (int t2 = 0; t2 < 4; ++t2) o[t2][g] *= a;
        }

        // ---- P: C-layout -> A-layout via per-wave LDS ----
        _Float16* pw = &pb[wv * 512];
        for (int g = 0; g < 8; ++g) {
            int row = off + g;
            pw[row * 32 + l16]      = (_Float16)s[0][g];
            pw[row * 32 + 16 + l16] = (_Float16)s[1][g];
        }
        asm volatile("s_wait_dscnt 0x0" ::: "memory");
        v16h ap = ld_frag(&pw[l16 * 32], off);

        // ---- O += P * V  (V B-frags gathered from row-major LDS tile) ----
        for (int t = 0; t < 4; ++t) {
            HV16 u;
            int dd = t * 16 + l16;
#pragma unroll
            for (int e = 0; e < 16; ++e) {
                int kk = ((e < 8) ? e : e + 8) + off;
                u.h[e] = vs[kk * 64 + dd];
            }
            o[t] = wmma16(ap, u.v, o[t]);
        }
    }

    for (int g = 0; g < 8; ++g) {
        float inv = 1.0f / lrow[g];
        int row = qr + off + g;
        _Float16* op = &oh[((size_t)b * N_ + row) * D_ + h * DH_ + l16];
        for (int t = 0; t < 4; ++t) op[t * 16] = (_Float16)(o[t][g] * inv);
    }
}

// ---------------- output projection: out = O @ Wout + b_out (fp32) ----------------
__global__ __launch_bounds__(128)
void gemm_out_kernel(const _Float16* __restrict__ ohh,
                     const _Float16* __restrict__ wouth,
                     const float* __restrict__ bout,
                     float* __restrict__ out) {
    __shared__ _Float16 wt[64 * 32];
    const int tid = threadIdx.x, lane = tid & 31, wv = tid >> 5;
    const int l16 = lane & 15, off = (lane < 16) ? 0 : 8;
    const int row0 = blockIdx.x * 64, n0 = blockIdx.y * 64;
    const _Float16* abase = ohh + (size_t)(row0 + wv * 16 + l16) * D_;

    v8f c[4]; for (int t = 0; t < 4; ++t) c[t] = {};
    for (int k0 = 0; k0 < D_; k0 += 32) {
        __syncthreads();
        for (int it = tid; it < 256; it += 128) {
            int kk = it >> 3, cc = (it & 7) * 8;
            const _Float16* wp = &wouth[(size_t)(k0 + kk) * D_ + n0 + cc];
            for (int q = 0; q < 8; ++q) wt[(cc + q) * 32 + kk] = wp[q];
        }
        __syncthreads();
        v16h a = ld_frag(abase + k0, off);
        for (int t = 0; t < 4; ++t)
            c[t] = wmma16(a, ld_frag(&wt[(t * 16 + l16) * 32], off), c[t]);
    }
    for (int t = 0; t < 4; ++t) {
        int col = n0 + t * 16 + l16;
        float bb = bout[col];
        for (int g = 0; g < 8; ++g) {
            int row = row0 + wv * 16 + off + g;
            out[(size_t)row * D_ + col] = c[t][g] + bb;
        }
    }
}

// ---------------- host launcher ----------------
extern "C" void kernel_launch(void* const* d_in, const int* in_sizes, int n_in,
                              void* d_out, int out_size, void* d_ws, size_t ws_size,
                              hipStream_t stream) {
    const float* x    = (const float*)d_in[0];   // (4,2048,1024)
    const float* xl   = (const float*)d_in[1];   // (4,512,2,64)
    const float* bias = (const float*)d_in[2];   // (16,2048,2560)
    const float* Wq   = (const float*)d_in[3];   // (1024,1024)
    const float* Wkv  = (const float*)d_in[4];   // (1024,128)
    const float* Wout = (const float*)d_in[5];   // (1024,1024)
    const float* bout = (const float*)d_in[6];   // (1024,)

    float* out    = (float*)d_out;               // (4,2048,1024) flat
    float* out_xl = out + (size_t)NR_ * D_;      // (4,512,2,64) flat

    char* ws = (char*)d_ws;
    size_t o = 0;
    _Float16* xh    = (_Float16*)(ws + o); o += (size_t)NR_ * D_ * 2;
    _Float16* wqh   = (_Float16*)(ws + o); o += (size_t)D_ * D_ * 2;
    _Float16* wkvh  = (_Float16*)(ws + o); o += (size_t)D_ * 128 * 2;
    _Float16* wouth = (_Float16*)(ws + o); o += (size_t)D_ * D_ * 2;
    _Float16* qh    = (_Float16*)(ws + o); o += (size_t)B_ * H_ * N_ * DH_ * 2;
    _Float16* kh    = (_Float16*)(ws + o); o += (size_t)B_ * J_ * DH_ * 2;
    _Float16* vh    = (_Float16*)(ws + o); o += (size_t)B_ * J_ * DH_ * 2;
    _Float16* ohh   = (_Float16*)(ws + o); o += (size_t)NR_ * D_ * 2;
    (void)ws_size; (void)in_sizes; (void)n_in; (void)out_size;

    int nx = NR_ * D_;
    cvt_f16_kernel<<<(nx + 255) / 256, 256, 0, stream>>>(x, xh, nx);
    cvt_f16_kernel<<<(D_ * D_ + 255) / 256, 256, 0, stream>>>(Wq, wqh, D_ * D_);
    cvt_f16_kernel<<<(D_ * 128 + 255) / 256, 256, 0, stream>>>(Wkv, wkvh, D_ * 128);
    cvt_f16_kernel<<<(D_ * D_ + 255) / 256, 256, 0, stream>>>(Wout, wouth, D_ * D_);
    xl_splice_kernel<<<(B_ * M_ * DH_ + 255) / 256, 256, 0, stream>>>(xl, kh, vh);

    gemm_q_kernel<<<dim3(NR_ / 64, D_ / 64), 128, 0, stream>>>(xh, wqh, qh);
    gemm_kv_kernel<<<dim3(NR_ / 64), 128, 0, stream>>>(xh, wkvh, kh, vh, out_xl);
    attn_kernel<<<dim3(N_ / 64, H_, B_), 128, 0, stream>>>(qh, kh, vh, bias, ohh);
    gemm_out_kernel<<<dim3(NR_ / 64, D_ / 64), 128, 0, stream>>>(ohh, wouth, bout, out);
}
